// SegmentPoolingWithPosEnc_84464826843887
// MI455X (gfx1250) — matrix operand order repaired and checked by reading, not compiled, not run on hardware
//
#include <hip/hip_runtime.h>
#include <hip/hip_bf16.h>

typedef __attribute__((ext_vector_type(16))) _Float16 v16h;
typedef __attribute__((ext_vector_type(8)))  float    v8f;

// Problem dims (fixed by the reference)
#define BB   16
#define NN   4096
#define CC   384
#define KK   256
#define NF   16        // NFREQ
#define F2   32        // 2*NFREQ
#define NTIL 24        // CC/16 channel tiles

// Output layout (flat float32 elements, concatenated in return order)
#define SP_OFF  0L
#define SP_LEN  ((long)BB * KK * CC)             // 1,572,864
#define OCC_OFF (SP_OFF + SP_LEN)
#define OCC_LEN ((long)BB * KK)                  // 4,096
#define A_OFF   (OCC_OFF + OCC_LEN)
#define A_LEN   ((long)BB * NN * KK)             // 16,777,216
#define P_OFF   (A_OFF + A_LEN)
#define P_LEN   ((long)BB * NN)                  // 65,536
#define SEG_OFF (P_OFF + P_LEN)                  // seg_len written as float

#define EPSF    1e-8f
#define LN_EPSF 1e-5f

// ---------------------------------------------------------------------------
// Kernel 0: zero the accumulator regions (s_parent + occ)
// ---------------------------------------------------------------------------
__global__ void k_zero(float* __restrict__ out, long n) {
    long i = (long)blockIdx.x * blockDim.x + threadIdx.x;
    if (i < n) out[i] = 0.0f;
}

// ---------------------------------------------------------------------------
// Kernel 1: run-length encoding via binary search on the sorted a_idx row.
// One block (256 threads) per batch row. Produces pos01 and seg_len.
// ---------------------------------------------------------------------------
__global__ __launch_bounds__(256) void k_segments(const int* __restrict__ aidx,
                                                  const unsigned char* __restrict__ nmask,
                                                  float* __restrict__ out) {
    __shared__ int sa[NN];
    __shared__ int s_nv;
    const int b = blockIdx.x;
    const int tid = threadIdx.x;

    if (tid == 0) s_nv = 0;
    // zero this row's seg_len slots (blockDim == KK == 256)
    out[SEG_OFF + (long)b * KK + tid] = 0.0f;
    __syncthreads();

    int cnt = 0;
    for (int i = tid; i < NN; i += 256) {
        sa[i] = aidx[(long)b * NN + i];
        cnt += nmask[(long)b * NN + i] ? 1 : 0;
    }
    atomicAdd(&s_nv, cnt);
    __syncthreads();
    const int nv = s_nv;   // valid tokens form a prefix (reference assumption)

    for (int n = tid; n < NN; n += 256) {
        float p = 0.0f;
        if (n < nv) {
            const int v = sa[n];
            // lower bound in [0, nv)
            int lo = 0, hi = nv;
            while (lo < hi) { int mid = (lo + hi) >> 1; if (sa[mid] <  v) lo = mid + 1; else hi = mid; }
            const int st = lo;
            // upper bound in [st, nv)
            lo = st; hi = nv;
            while (lo < hi) { int mid = (lo + hi) >> 1; if (sa[mid] <= v) lo = mid + 1; else hi = mid; }
            const int en = lo;
            const int L = en - st;
            if (L > 1) p = (float)(n - st) / ((float)(L - 1) + EPSF);
            // run start -> scatter seg_len (runs unique per k for sorted a_idx)
            if (n == st && v >= 0 && v < KK)
                out[SEG_OFF + (long)b * KK + v] = (float)L;
        }
        out[P_OFF + (long)b * NN + n] = p;
    }
}

// ---------------------------------------------------------------------------
// Kernel 2: convert W_proj [C][2F] f32 -> f16 B-tiles in FRAGMENT order:
//   Wh[t*512 + lane*16 + j]  with lane = ((k>>3)&1)*16 + n, j = (k>>4)*8 + (k&7), c = t*16+n
// so each lane's 16 B-fragment halves are 32 contiguous bytes.
// ---------------------------------------------------------------------------
__global__ void k_convW(const float* __restrict__ W, _Float16* __restrict__ Wh) {
    int i = blockIdx.x * blockDim.x + threadIdx.x;
    if (i >= NTIL * F2 * 16) return;
    const int t    = i >> 9;
    const int r    = i & 511;
    const int lane = r >> 4;
    const int j    = r & 15;
    const int n    = lane & 15;
    const int k    = (j >> 3) * 16 + (lane >> 4) * 8 + (j & 7);
    Wh[i] = (_Float16)W[(t * 16 + n) * F2 + k];
}

// ---------------------------------------------------------------------------
// Kernel 3: positional encoding via WMMA (K=32 == 2*NFREQ, one WMMA/tile),
// masked LayerNorm over C, s_aug = s + pos_emb, atomic segment-sum pooling.
// Block = 256 threads (8 waves). Each wave owns a 16-token M-tile.
// LDS holds A and B data in fragment order -> ds_load_b128 fragment loads.
// ---------------------------------------------------------------------------
__global__ __launch_bounds__(256) void k_posenc_pool(const float* __restrict__ s,
                                                     const unsigned char* __restrict__ nmask,
                                                     const int* __restrict__ aidx,
                                                     const float* __restrict__ gamma,
                                                     const float* __restrict__ beta,
                                                     const _Float16* __restrict__ Wh,
                                                     const float* __restrict__ pos01,
                                                     float* __restrict__ sparent) {
    __shared__ __align__(32) _Float16 sFeat[8 * 512];        // 8 KB, A-fragment order
    __shared__ __align__(32) _Float16 sW[NTIL * 512];        // 24 KB, B-fragment order

    const int tid  = threadIdx.x;
    const int b    = blockIdx.x >> 5;             // 32 blocks per batch row
    const int tok0 = (blockIdx.x & 31) * 128;

    // ---- phase 1: sin/cos features for 128 tokens into LDS in A-frag order.
    // 16-bit A 16x32 layout (ISA 7.12.2): lane = kh*16 + m holds halves j,
    // with k = (r<4?0:16) + kh*8 + (r&3)*2 + p, r=j>>1, p=j&1.
    // Inverse (given k): hi=k>=16, f=k&15, kh=f>=8, r=hi*4+((f&7)>>1), p=f&1.
    {
        const int tk = tid >> 1;                  // token within block, 0..127
        const int hc = tid & 1;                   // 0 = sin (k=f), 1 = cos (k=16+f)
        const int wv = tk >> 4;
        const int m  = tk & 15;
        float p = pos01[(long)b * NN + tok0 + tk];
        p = fminf(fmaxf(p, 0.0f), 1.0f);
        const float base = 6.28318530717958647692f * p;
#pragma unroll
        for (int f = 0; f < NF; ++f) {
            const float ang = base * (float)(1 << f);
            const float v = hc ? __cosf(ang) : __sinf(ang);
            const int kh = (f >= 8) ? 1 : 0;
            const int r  = hc * 4 + ((f & 7) >> 1);
            const int j  = r * 2 + (f & 1);
            sFeat[wv * 512 + (kh * 16 + m) * 16 + j] = (_Float16)v;
        }
    }
    // ---- phase 2: stage f16 weight tiles into LDS (vectorized int4 copy) ----
    {
        const int4* __restrict__ src = (const int4*)Wh;
        int4* dst = (int4*)sW;
        const int n16 = NTIL * 512 * 2 / 16;      // 1536 int4s
        for (int i = tid; i < n16; i += 256) dst[i] = src[i];
    }
    __syncthreads();

    const int wv   = tid >> 5;                    // wave id 0..7
    const int lane = tid & 31;
    const int mcol = lane & 15;                   // N column of D
    const int kh   = lane >> 4;                   // half-group select
    const int tokBase = tok0 + wv * 16;

    // ---- A fragment: one aligned 32-byte LDS read ----
    const v16h afr = *(const v16h*)&sFeat[wv * 512 + lane * 16];

    // per-token metadata: lane holds rows m = kh*8 + v, v = 0..7
    float mtok[8]; int kq[8];
#pragma unroll
    for (int v = 0; v < 8; ++v) {
        const int tok = tokBase + kh * 8 + v;
        const long gi = (long)b * NN + tok;
        mtok[v] = nmask[gi] ? 1.0f : 0.0f;
        int a = aidx[gi];
        kq[v] = min(max(a, 0), KK - 1);
    }

    // ---- pass 1: 24 WMMAs, accumulate masked sum / sumsq per token ----
    float sum[8], sq[8];
#pragma unroll
    for (int v = 0; v < 8; ++v) { sum[v] = 0.0f; sq[v] = 0.0f; }

    for (int t = 0; t < NTIL; ++t) {
        const v16h bfr = *(const v16h*)&sW[t * 512 + lane * 16];
        v8f cz = {};
        v8f c = __builtin_amdgcn_wmma_f32_16x16x32_f16(
            false, afr, false, bfr, (short)0, cz, false, false);
#pragma unroll
        for (int v = 0; v < 8; ++v) {
            const float x = c[v] * mtok[v];
            sum[v] += x; sq[v] += x * x;
        }
    }

    // ---- cross-lane reduce within each 16-lane group (same token rows) ----
    float muv[8], rinv[8];
#pragma unroll
    for (int v = 0; v < 8; ++v) {
        float s1 = sum[v], s2 = sq[v];
        for (int off = 1; off < 16; off <<= 1) {
            s1 += __shfl_xor(s1, off, 32);
            s2 += __shfl_xor(s2, off, 32);
        }
        const float mu  = s1 * (1.0f / (float)CC);
        const float var = s2 * (1.0f / (float)CC) - mu * mu;
        muv[v]  = mu;
        rinv[v] = rsqrtf(var + LN_EPSF);
    }

    // ---- pass 2: recompute tiles, LayerNorm, add s, pool into s_parent ----
    for (int t = 0; t < NTIL; ++t) {
        const v16h bfr = *(const v16h*)&sW[t * 512 + lane * 16];
        v8f cz = {};
        v8f c = __builtin_amdgcn_wmma_f32_16x16x32_f16(
            false, afr, false, bfr, (short)0, cz, false, false);

        const int ch = t * 16 + mcol;
        const float g  = gamma[ch];
        const float be = beta[ch];
#pragma unroll
        for (int v = 0; v < 8; ++v) {
            const int tok = tokBase + kh * 8 + v;
            const float x  = c[v] * mtok[v];
            const float pe = (x - muv[v]) * rinv[v] * g + be;
            const float sa = s[((long)b * NN + tok) * CC + ch] + pe;
            if (mtok[v] != 0.0f)
                atomicAdd(&sparent[((long)(b * KK + kq[v])) * CC + ch], sa);
        }
    }
}

// ---------------------------------------------------------------------------
// Kernel 4: one-hot A matrix (coalesced float4 stores) + occ accumulation
// ---------------------------------------------------------------------------
__global__ void k_amat(const int* __restrict__ aidx,
                       const unsigned char* __restrict__ nmask,
                       float* __restrict__ out) {
    const long e4 = (long)blockIdx.x * blockDim.x + threadIdx.x;
    const long tot = A_LEN / 4;
    if (e4 >= tot) return;
    const int  k4  = (int)(e4 & 63);              // float4 index within K=256
    const long row = e4 >> 6;                     // b*N + n
    const int  a   = aidx[row];
    const float m  = nmask[row] ? 1.0f : 0.0f;
    const int  kq  = min(max(a, 0), KK - 1);

    float4 val = make_float4(0.f, 0.f, 0.f, 0.f);
    const int base = k4 * 4;
    if (kq >= base && kq < base + 4) {
        ((float*)&val)[kq - base] = m;
        if (m != 0.0f)
            atomicAdd(&out[OCC_OFF + (row >> 12) * KK + kq], m);   // row>>12 == b
    }
    ((float4*)(out + A_OFF))[e4] = val;
}

// ---------------------------------------------------------------------------
// Kernel 5: finalize s_parent: divide by max(occ,eps), apply mask_parent
// ---------------------------------------------------------------------------
__global__ void k_finalize(const unsigned char* __restrict__ maskp,
                           float* __restrict__ out) {
    const long e = (long)blockIdx.x * blockDim.x + threadIdx.x;
    if (e >= SP_LEN) return;
    const long bk = e / CC;
    const float occ = out[OCC_OFF + bk];
    const float mp  = maskp[bk] ? 1.0f : 0.0f;
    out[e] = out[e] * mp / fmaxf(occ, EPSF);
}

// ---------------------------------------------------------------------------
extern "C" void kernel_launch(void* const* d_in, const int* in_sizes, int n_in,
                              void* d_out, int out_size, void* d_ws, size_t ws_size,
                              hipStream_t stream) {
    const float*         s     = (const float*)d_in[0];
    const unsigned char* nmask = (const unsigned char*)d_in[1];
    const int*           aidx  = (const int*)d_in[2];
    const unsigned char* maskp = (const unsigned char*)d_in[3];
    const float*         Wproj = (const float*)d_in[4];
    const float*         gam   = (const float*)d_in[5];
    const float*         bet   = (const float*)d_in[6];
    float*               out   = (float*)d_out;
    _Float16*            Wh    = (_Float16*)d_ws;  // 24 KB scratch

    // 0: zero accumulators (s_parent + occ)
    {
        const long n = SP_LEN + OCC_LEN;
        k_zero<<<(int)((n + 255) / 256), 256, 0, stream>>>(out, n);
    }
    // 1: pos01 + seg_len
    k_segments<<<BB, 256, 0, stream>>>(aidx, nmask, out);
    // 2: f16 weight tiles (fragment order)
    k_convW<<<(NTIL * F2 * 16 + 255) / 256, 256, 0, stream>>>(Wproj, Wh);
    // 3: WMMA pos-encode + LN + pooling accumulation
    k_posenc_pool<<<(BB * NN) / 128, 256, 0, stream>>>(
        s, nmask, aidx, gam, bet, Wh, out + P_OFF, out + SP_OFF);
    // 4: A one-hot + occ
    {
        const long n4 = A_LEN / 4;
        k_amat<<<(int)((n4 + 255) / 256), 256, 0, stream>>>(aidx, nmask, out);
    }
    // 5: finalize s_parent
    k_finalize<<<(int)((SP_LEN + 255) / 256), 256, 0, stream>>>(maskp, out);
}
